// _NonLocalBlockND_987842478424
// MI455X (gfx1250) — matrix operand order repaired and checked by reading, not compiled, not run on hardware
//
#include <hip/hip_runtime.h>
#include <hip/hip_bf16.h>

typedef __attribute__((ext_vector_type(16))) __bf16 v16bf;
typedef __attribute__((ext_vector_type(8)))  __bf16 v8bf;
typedef __attribute__((ext_vector_type(8)))  float  v8f;

#define DEV_ __device__ __forceinline__

constexpr int kB = 16, kC = 256, kD = 128, kW = 64;
constexpr int kN = 4096;   // H*W
constexpr int kM = 1024;   // (H/2)*(W/2)
constexpr int GRAM_WGS = 64;

DEV_ v16bf cat8(v8bf lo, v8bf hi) {
  return __builtin_shufflevector(lo, hi, 0,1,2,3,4,5,6,7,8,9,10,11,12,13,14,15);
}
// A-fragment 16x32 bf16 (ISA 7.12.2): lanes 0-15 row=l K={kb..kb+7, kb+16..kb+23},
// lanes 16-31 row=l K={kb+8..kb+15, kb+24..kb+31}. src row-major [row][k], stride ld.
DEV_ v16bf load_afrag(const __bf16* src, int ld, int row, int kbase, int lane) {
  int half = lane >> 4, l = lane & 15;
  const __bf16* p = src + (size_t)(row + l) * ld + kbase + half * 8;
  return cat8(*(const v8bf*)p, *(const v8bf*)(p + 16));
}
// B-fragment 32x16 bf16: lane col = cb+l, K = {kb+half*8..+7, kb+16+half*8..+7} (same as A).
// src layout [col][k], stride ld.
DEV_ v16bf load_bfrag(const __bf16* src, int ld, int col, int kbase, int lane) {
  int half = lane >> 4, l = lane & 15;
  const __bf16* p = src + (size_t)(col + l) * ld + kbase + half * 8;
  return cat8(*(const v8bf*)p, *(const v8bf*)(p + 16));
}
DEV_ v8f wmma_bf16(v16bf a, v16bf b, v8f c) {
  return __builtin_amdgcn_wmma_f32_16x16x32_bf16(false, a, false, b, (short)0, c, false, false);
}
DEV_ v8f vzero() { v8f z = {0.f,0.f,0.f,0.f,0.f,0.f,0.f,0.f}; return z; }

// CDNA5 async global->LDS DMA (GLOBAL_LOAD_ASYNC_TO_LDS_B128, ASYNCcnt-tracked).
// VDST operand carries the LDS byte address; low 32 bits of a generic LDS pointer
// are the LDS byte address (flat-addressing truncation rule).
DEV_ void async_b128(__bf16* lds_dst, const __bf16* gsrc) {
  unsigned lds_addr = (unsigned)(size_t)lds_dst;
  asm volatile("global_load_async_to_lds_b128 %0, %1, off"
               :: "v"(lds_addr), "v"(gsrc) : "memory");
}
DEV_ void wait_async0() { asm volatile("s_wait_asynccnt 0" ::: "memory"); }

// ---------------- weight / layout prep ----------------

__global__ void k_cvt_bf16(const float* __restrict__ s, __bf16* __restrict__ d, int n) {
  int i = blockIdx.x * 256 + threadIdx.x;
  if (i < n) d[i] = (__bf16)s[i];
}

// x[b][c][n] -> thetaT[b][n][c] bf16 (LDS-tiled transpose)
__global__ void k_theta_transpose(const float* __restrict__ x, __bf16* __restrict__ thetaT) {
  __shared__ __bf16 tile[32][72];
  int b = blockIdx.z, c0 = blockIdx.y * 32, n0 = blockIdx.x * 64;
  int t = threadIdx.x;
  {
    int nl = t & 63, cg = t >> 6;
    const float* xp = x + ((size_t)b * kC + c0) * kN + n0 + nl;
    #pragma unroll
    for (int i = 0; i < 8; i++) {
      int cl = cg * 8 + i;
      tile[cl][nl] = (__bf16)xp[(size_t)cl * kN];
    }
  }
  __syncthreads();
  {
    int cl = t & 31, ng = t >> 5;
    __bf16* op = thetaT + ((size_t)b * kN + n0) * kC + c0 + cl;
    #pragma unroll
    for (int i = 0; i < 8; i++) {
      int nl = ng * 8 + i;
      op[(size_t)nl * kC] = tile[cl][nl];
    }
  }
}

// 2x2 maxpool of x then transpose -> phiT[b][m][c] bf16
__global__ void k_pool_phi(const float* __restrict__ x, __bf16* __restrict__ phiT) {
  __shared__ __bf16 tile[32][72];
  int b = blockIdx.z, c0 = blockIdx.y * 32, m0 = blockIdx.x * 64;
  int t = threadIdx.x;
  {
    int ml = t & 63, cg = t >> 6;
    int m = m0 + ml, ph = m >> 5, pw = m & 31;
    const float* xp = x + ((size_t)b * kC + c0) * kN + (size_t)(ph * 2) * kW + pw * 2;
    #pragma unroll
    for (int i = 0; i < 8; i++) {
      int cl = cg * 8 + i;
      const float* q = xp + (size_t)cl * kN;
      float v = fmaxf(fmaxf(q[0], q[1]), fmaxf(q[kW], q[kW + 1]));
      tile[cl][ml] = (__bf16)v;
    }
  }
  __syncthreads();
  {
    int cl = t & 31, mg = t >> 5;
    __bf16* op = phiT + ((size_t)b * kM + m0) * kC + c0 + cl;
    #pragma unroll
    for (int i = 0; i < 8; i++) {
      int ml = mg * 8 + i;
      op[(size_t)ml * kC] = tile[cl][ml];
    }
  }
}

// ---------------- g branch: [N,C]@[C,D] WMMA GEMM + fused 2x2 maxpool ----------------
// grid (kN/128, kB). Weights staged once per WG in LDS; pooling staging aliases it.
__global__ void __launch_bounds__(256)
k_g_gemm_pool(const __bf16* __restrict__ thetaT, const __bf16* __restrict__ g_wb,
              const float* __restrict__ g_b, __bf16* __restrict__ gx) {
  __shared__ __bf16 smem[128 * 264];          // 67.6 KB: weights [128][264], then gs [128][138]
  __bf16* wl = smem;
  __bf16* gs = smem;
  int t = threadIdx.x, lane = t & 31, wv = t >> 5;
  int half = lane >> 4, l15 = lane & 15;
  int b = blockIdx.y, n0 = blockIdx.x * 128;

  { // cooperative weight stage: g_wb [128][256] -> wl [128][264]
    int row = t >> 1, hg = t & 1;
    const __bf16* sp = g_wb + (size_t)row * kC + hg * 128;
    __bf16* dp = wl + (size_t)row * 264 + hg * 128;
    #pragma unroll
    for (int i = 0; i < 16; i++)
      *(v8bf*)(dp + i * 8) = *(const v8bf*)(sp + i * 8);
  }

  v16bf ath[8];
  {
    const __bf16* tp = thetaT + ((size_t)b * kN + n0 + wv * 16 + l15) * kC;
    #pragma unroll
    for (int k = 0; k < 8; k++) {
      const __bf16* p = tp + k * 32 + half * 8;
      ath[k] = cat8(*(const v8bf*)p, *(const v8bf*)(p + 16));
    }
  }
  __syncthreads();

  v8f acc[8];
  #pragma unroll
  for (int i = 0; i < 8; i++) acc[i] = vzero();
  #pragma unroll
  for (int k = 0; k < 8; k++) {
    #pragma unroll
    for (int dt = 0; dt < 8; dt++) {
      v16bf bw = load_bfrag(wl, 264, dt * 16, k * 32, lane);
      acc[dt] = wmma_bf16(ath[k], bw, acc[dt]);
    }
  }
  __syncthreads();   // all waves done reading weights before gs overwrites the aliased LDS
  #pragma unroll
  for (int dt = 0; dt < 8; dt++)
    #pragma unroll
    for (int r = 0; r < 8; r++)
      gs[(size_t)(wv * 16 + r + 8 * half) * 138 + dt * 16 + l15] = (__bf16)acc[dt][r];
  __syncthreads();
  int ph = blockIdx.x;
  int pw = t & 31, dg = t >> 5;
  #pragma unroll
  for (int i = 0; i < 16; i++) {
    int d = dg * 16 + i;
    float v0 = (float)gs[(size_t)(2 * pw) * 138 + d];
    float v1 = (float)gs[(size_t)(2 * pw + 1) * 138 + d];
    float v2 = (float)gs[(size_t)(64 + 2 * pw) * 138 + d];
    float v3 = (float)gs[(size_t)(65 + 2 * pw) * 138 + d];
    float mx = fmaxf(fmaxf(v0, v1), fmaxf(v2, v3)) + g_b[d];
    gx[((size_t)b * kD + d) * kM + (size_t)ph * 32 + pw] = (__bf16)mx;
  }
}

// ---------------- flash attention: softmax(theta @ phi) @ g, streamed over M ----------------
// Double-buffered chunks, HBM->LDS via async DMA overlapped with WMMA.
__global__ void __launch_bounds__(256)
k_flash(const __bf16* __restrict__ thetaT, const __bf16* __restrict__ phiT,
        const __bf16* __restrict__ gx, __bf16* __restrict__ y) {
  __shared__ __bf16 phl[2][32 * 264];   // phi chunk [32 m][264]  x2
  __shared__ __bf16 gl[2][128 * 40];    // g chunk   [128 d][40]  x2
  __shared__ __bf16 Pl[8 * 16 * 40];    // per-wave probs [16][40]
  int t = threadIdx.x, lane = t & 31, wv = t >> 5;
  int half = lane >> 4, l15 = lane & 15;
  int b = blockIdx.y, n0 = blockIdx.x * 128;

  // per-thread chunk-copy addressing
  int prow = t >> 3, pcg = t & 7;                  // phi: 32B per thread x2
  const __bf16* phi_src0 = phiT + ((size_t)b * kM + prow) * kC + pcg * 32;
  int gd = t >> 1, ghg = t & 1;                    // g: 32B per thread
  const __bf16* g_src0 = gx + ((size_t)b * kD + gd) * kM + ghg * 16;

  v16bf ath[8];
  {
    const __bf16* tp = thetaT + ((size_t)b * kN + n0 + wv * 16 + l15) * kC;
    #pragma unroll
    for (int k = 0; k < 8; k++) {
      const __bf16* p = tp + k * 32 + half * 8;
      ath[k] = cat8(*(const v8bf*)p, *(const v8bf*)(p + 16));
    }
  }
  v8f acc[8];
  #pragma unroll
  for (int i = 0; i < 8; i++) acc[i] = vzero();
  float mrow[8], lrow[8];
  #pragma unroll
  for (int r = 0; r < 8; r++) { mrow[r] = -3.0e38f; lrow[r] = 0.f; }
  __bf16* Pw = Pl + wv * 16 * 40;

  // prologue: async-load chunk 0 into buffer 0
  {
    __bf16* pd = phl[0] + prow * 264 + pcg * 32;
    async_b128(pd, phi_src0);
    async_b128(pd + 8, phi_src0 + 8);
    async_b128(pd + 16, phi_src0 + 16);
    async_b128(pd + 24, phi_src0 + 24);
    __bf16* gdp = gl[0] + gd * 40 + ghg * 16;
    async_b128(gdp, g_src0);
    async_b128(gdp + 8, g_src0 + 8);
  }

  for (int mc = 0; mc < 32; mc++) {
    int cur = mc & 1;
    wait_async0();           // this wave's chunk-mc DMAs are in LDS
    __syncthreads();         // ...and everyone else's too
    if (mc < 31) {           // prefetch chunk mc+1 into the other buffer
      int m1 = (mc + 1) * 32;
      __bf16* pd = phl[cur ^ 1] + prow * 264 + pcg * 32;
      const __bf16* ps = phi_src0 + (size_t)m1 * kC;
      async_b128(pd, ps);
      async_b128(pd + 8, ps + 8);
      async_b128(pd + 16, ps + 16);
      async_b128(pd + 24, ps + 24);
      __bf16* gdp = gl[cur ^ 1] + gd * 40 + ghg * 16;
      const __bf16* gsp = g_src0 + m1;
      async_b128(gdp, gsp);
      async_b128(gdp + 8, gsp + 8);
    }

    const __bf16* phc = phl[cur];
    const __bf16* glc = gl[cur];
    v8f s0 = vzero(), s1 = vzero();
    #pragma unroll
    for (int k = 0; k < 8; k++) {
      v16bf b0 = load_bfrag(phc, 264, 0, k * 32, lane);
      v16bf b1 = load_bfrag(phc, 264, 16, k * 32, lane);
      s0 = wmma_bf16(ath[k], b0, s0);
      s1 = wmma_bf16(ath[k], b1, s1);
    }
    // online softmax; C/D layout: row = r + 8*half spread over 16 lanes (col = l15)
    #pragma unroll
    for (int r = 0; r < 8; r++) {
      float cm = fmaxf(s0[r], s1[r]);
      cm = fmaxf(cm, __shfl_xor(cm, 1, 32));
      cm = fmaxf(cm, __shfl_xor(cm, 2, 32));
      cm = fmaxf(cm, __shfl_xor(cm, 4, 32));
      cm = fmaxf(cm, __shfl_xor(cm, 8, 32));
      float nm = fmaxf(mrow[r], cm);
      float sc = __expf(mrow[r] - nm);
      float p0 = __expf(s0[r] - nm);
      float p1 = __expf(s1[r] - nm);
      Pw[(r + 8 * half) * 40 + l15]      = (__bf16)p0;
      Pw[(r + 8 * half) * 40 + 16 + l15] = (__bf16)p1;
      float rs = p0 + p1;
      rs += __shfl_xor(rs, 1, 32);
      rs += __shfl_xor(rs, 2, 32);
      rs += __shfl_xor(rs, 4, 32);
      rs += __shfl_xor(rs, 8, 32);
      lrow[r] = lrow[r] * sc + rs;
      mrow[r] = nm;
      #pragma unroll
      for (int dt = 0; dt < 8; dt++) acc[dt][r] *= sc;
    }
    asm volatile("s_wait_dscnt 0" ::: "memory"); // P writes visible before A-frag reload
    v16bf pa = load_afrag(Pw, 40, 0, 0, lane);
    #pragma unroll
    for (int dt = 0; dt < 8; dt++) {
      v16bf bg = load_bfrag(glc, 40, dt * 16, 0, lane);
      acc[dt] = wmma_bf16(pa, bg, acc[dt]);
    }
    // no trailing barrier: next iteration's wait+barrier guards buffer reuse
  }
  __bf16* yp = y + ((size_t)b * kN + n0 + wv * 16) * kD;
  #pragma unroll
  for (int dt = 0; dt < 8; dt++)
    #pragma unroll
    for (int r = 0; r < 8; r++) {
      float v = acc[dt][r] / lrow[r];
      yp[(size_t)(r + 8 * half) * kD + dt * 16 + l15] = (__bf16)v;
    }
}

// ---------------- Gram matrix G = Y^T Y (128x128) + column sums of Y ----------------
__global__ void __launch_bounds__(256)
k_gram(const __bf16* __restrict__ y, float* __restrict__ pG, float* __restrict__ pS) {
  __shared__ __bf16 yT[128 * 40];     // transposed tile [128 d][40]
  int t = threadIdx.x, lane = t & 31, wv = t >> 5;
  int half = lane >> 4, l15 = lane & 15;
  int rl = t & 31, dc = t >> 5;       // rl: row-in-tile, dc(==wv): d-chunk of 16
  size_t slab = (size_t)blockIdx.x * 1024;
  v8f g[8];
  #pragma unroll
  for (int i = 0; i < 8; i++) g[i] = vzero();
  float sum16[16];
  #pragma unroll
  for (int i = 0; i < 16; i++) sum16[i] = 0.f;

  for (int tile = 0; tile < 32; tile++) {
    size_t row0 = slab + (size_t)tile * 32;
    const __bf16* sp = y + (row0 + rl) * kD + dc * 16;
    v8bf a  = *(const v8bf*)(sp + 0);
    v8bf bq = *(const v8bf*)(sp + 8);
    #pragma unroll
    for (int i = 0; i < 8; i++) {
      yT[(dc * 16 + i) * 40 + rl]     = a[i];
      yT[(dc * 16 + 8 + i) * 40 + rl] = bq[i];
      sum16[i]     += (float)a[i];
      sum16[8 + i] += (float)bq[i];
    }
    __syncthreads();
    v16bf af = load_afrag(yT, 40, wv * 16, 0, lane);
    #pragma unroll
    for (int dt = 0; dt < 8; dt++) {
      v16bf bf = load_bfrag(yT, 40, dt * 16, 0, lane);
      g[dt] = wmma_bf16(af, bf, g[dt]);
    }
    __syncthreads();
  }
  float* pGo = pG + (size_t)blockIdx.x * 16384;
  #pragma unroll
  for (int dt = 0; dt < 8; dt++)
    #pragma unroll
    for (int r = 0; r < 8; r++)
      pGo[(size_t)(wv * 16 + r + 8 * half) * 128 + dt * 16 + l15] = g[dt][r];
  #pragma unroll
  for (int i = 0; i < 16; i++) {
    float v = sum16[i];
    v += __shfl_xor(v, 16, 32);
    v += __shfl_xor(v, 8, 32);
    v += __shfl_xor(v, 4, 32);
    v += __shfl_xor(v, 2, 32);
    v += __shfl_xor(v, 1, 32);
    if (lane == 0) pS[(size_t)blockIdx.x * 128 + dc * 16 + i] = v;
  }
}

__global__ void k_gram_reduce(const float* __restrict__ pG, const float* __restrict__ pS,
                              float* __restrict__ G, float* __restrict__ ysum) {
  int idx = blockIdx.x * 256 + threadIdx.x;
  float s = 0.f;
  for (int w = 0; w < GRAM_WGS; w++) s += pG[(size_t)w * 16384 + idx];
  G[idx] = s;
  if (blockIdx.x == 0 && threadIdx.x < 128) {
    float t2 = 0.f;
    for (int w = 0; w < GRAM_WGS; w++) t2 += pS[(size_t)w * 128 + threadIdx.x];
    ysum[threadIdx.x] = t2;
  }
}

// exact BN stats from Gram: one block, one channel per thread
__global__ void k_stats(const float* __restrict__ G, const float* __restrict__ ysum,
                        const float* __restrict__ w_w, const float* __restrict__ w_b,
                        const float* __restrict__ gamma, const float* __restrict__ beta,
                        float* __restrict__ scaleA, float* __restrict__ shiftF) {
  int c = threadIdx.x;
  const float cnt = (float)((size_t)kB * kN);
  const float* wc = w_w + (size_t)c * kD;
  float dot = 0.f;
  for (int d = 0; d < kD; d++) dot += wc[d] * ysum[d];
  float quad = 0.f;
  for (int d1 = 0; d1 < kD; d1++) {
    float inner = 0.f;
    const float* Gr = G + (size_t)d1 * kD;
    for (int d2 = 0; d2 < kD; d2++) inner += Gr[d2] * wc[d2];
    quad += wc[d1] * inner;
  }
  float wb = w_b[c];
  float mean = dot / cnt + wb;
  float e2 = (quad + 2.f * wb * dot) / cnt + wb * wb;
  float var = e2 - mean * mean;
  float sA = gamma[c] * rsqrtf(var + 1e-5f);
  scaleA[c] = sA;
  shiftF[c] = beta[c] + (wb - mean) * sA;
}

// ---------------- final: out = (y @ W^T)*scale + shift + x ----------------
__global__ void __launch_bounds__(256)
k_out(const __bf16* __restrict__ y, const __bf16* __restrict__ w_wb,
      const float* __restrict__ scaleA, const float* __restrict__ shiftF,
      const float* __restrict__ x, float* __restrict__ out) {
  __shared__ __bf16 wl[256 * 136];    // 69.6 KB weight cache [256 c][136]
  int t = threadIdx.x, lane = t & 31, wv = t >> 5;
  int half = lane >> 4, l15 = lane & 15;
  int b = blockIdx.y, n0 = blockIdx.x * 128;

  { // cooperative weight stage: w_wb [256][128] -> wl [256][136]
    const __bf16* sp = w_wb + (size_t)t * kD;
    __bf16* dp = wl + (size_t)t * 136;
    #pragma unroll
    for (int i = 0; i < 16; i++)
      *(v8bf*)(dp + i * 8) = *(const v8bf*)(sp + i * 8);
  }

  v16bf ay[4];
  const __bf16* yp = y + ((size_t)b * kN + n0 + wv * 16 + l15) * kD;
  #pragma unroll
  for (int k = 0; k < 4; k++) {
    const __bf16* p = yp + k * 32 + half * 8;
    ay[k] = cat8(*(const v8bf*)p, *(const v8bf*)(p + 16));
  }
  __syncthreads();

  #pragma unroll
  for (int pass = 0; pass < 2; pass++) {
    v8f acc[8];
    #pragma unroll
    for (int i = 0; i < 8; i++) acc[i] = vzero();
    #pragma unroll
    for (int k = 0; k < 4; k++) {
      #pragma unroll
      for (int ct = 0; ct < 8; ct++) {
        v16bf bw = load_bfrag(wl, 136, pass * 128 + ct * 16, k * 32, lane);
        acc[ct] = wmma_bf16(ay[k], bw, acc[ct]);
      }
    }
    #pragma unroll
    for (int ct = 0; ct < 8; ct++) {
      int c = pass * 128 + ct * 16 + l15;
      float sA = scaleA[c], sF = shiftF[c];
      size_t base = ((size_t)b * kC + c) * kN + n0 + wv * 16 + 8 * half;
      #pragma unroll
      for (int r = 0; r < 8; r++)
        out[base + r] = acc[ct][r] * sA + sF + x[base + r];
    }
  }
}

// ---------------- host ----------------

extern "C" void kernel_launch(void* const* d_in, const int* in_sizes, int n_in,
                              void* d_out, int out_size, void* d_ws, size_t ws_size,
                              hipStream_t stream) {
  const float* x     = (const float*)d_in[0];
  const float* g_w   = (const float*)d_in[1];
  const float* g_b   = (const float*)d_in[2];
  const float* w_w   = (const float*)d_in[3];
  const float* w_b   = (const float*)d_in[4];
  const float* gamma = (const float*)d_in[5];
  const float* beta  = (const float*)d_in[6];
  float* out = (float*)d_out;

  char* ws = (char*)d_ws;
  __bf16* thetaT = (__bf16*)(ws + 0);              // 33,554,432 B
  __bf16* phiT   = (__bf16*)(ws + 33554432);       //  8,388,608 B
  __bf16* gx     = (__bf16*)(ws + 41943040);       //  4,194,304 B
  __bf16* yb     = (__bf16*)(ws + 46137344);       // 16,777,216 B
  __bf16* g_wb   = (__bf16*)(ws + 62914560);       //     65,536 B
  __bf16* w_wb   = (__bf16*)(ws + 62980096);       //     65,536 B
  float*  pG     = (float*)(ws + 63045632);        //  4,194,304 B
  float*  pS     = (float*)(ws + 67239936);        //     32,768 B
  float*  G      = (float*)(ws + 67272704);        //     65,536 B
  float*  ysum   = (float*)(ws + 67338240);        //        512 B
  float*  scaleA = (float*)(ws + 67338752);        //      1,024 B
  float*  shiftF = (float*)(ws + 67339776);        //      1,024 B

  k_cvt_bf16<<<dim3(128), 256, 0, stream>>>(g_w, g_wb, kD * kC);
  k_cvt_bf16<<<dim3(128), 256, 0, stream>>>(w_w, w_wb, kC * kD);
  k_theta_transpose<<<dim3(kN / 64, kC / 32, kB), 256, 0, stream>>>(x, thetaT);
  k_pool_phi<<<dim3(kM / 64, kC / 32, kB), 256, 0, stream>>>(x, phiT);
  k_g_gemm_pool<<<dim3(kN / 128, kB), 256, 0, stream>>>(thetaT, g_wb, g_b, gx);
  k_flash<<<dim3(kN / 128, kB), 256, 0, stream>>>(thetaT, phiT, gx, yb);
  k_gram<<<dim3(GRAM_WGS), 256, 0, stream>>>(yb, pG, pS);
  k_gram_reduce<<<dim3(64), 256, 0, stream>>>(pG, pS, G, ysum);
  k_stats<<<dim3(1), 256, 0, stream>>>(G, ysum, w_w, w_b, gamma, beta, scaleA, shiftF);
  k_out<<<dim3(kN / 128, kB), 256, 0, stream>>>(yb, w_wb, scaleA, shiftF, x, out);
}
